// Block_41918880809530
// MI455X (gfx1250) — compile-verified, matrix operand order checked by reading, not run on hardware
//
#include <hip/hip_runtime.h>
#include <stdint.h>

#define Bb 4
#define Ss 2048
#define Dd 1024
#define Hh 16
#define HD 64
#define FFf 4096
#define Ee 8
#define Tt (Bb * Ss)        // 8192
#define CAPc (Tt / Ee)      // 1024
#define SCALE_INV 0.125f    // 1/sqrt(64)

typedef __attribute__((ext_vector_type(16))) _Float16 v16h;
typedef __attribute__((ext_vector_type(8)))  float    v8f;
typedef __attribute__((ext_vector_type(4)))  unsigned int v4u;
typedef __attribute__((ext_vector_type(8)))  int      v8i;
typedef __attribute__((ext_vector_type(4)))  int      v4i;

union F16x16 { v16h v; _Float16 h[16]; uint32_t u[8]; };
union F32x8  { v8f  v; float f[8]; };

__device__ __forceinline__ v8f wmma_f16(v16h a, v16h b, v8f c) {
  return __builtin_amdgcn_wmma_f32_16x16x32_f16(false, a, false, b, (short)0, c, false, false);
}

// K index of first half held in VGPR p for this lane (16-bit A/B fragment layout).
__device__ __forceinline__ int kmap(int p, int lane) {
  int k = (p < 4) ? (p << 1) : (16 + ((p - 4) << 1));
  return k + ((lane >> 4) << 3);
}

__device__ __forceinline__ float gelu_exact(float x) {
  return 0.5f * x * (1.0f + erff(x * 0.70710678118654752f));
}

// ---------------------------------------------------------------- TDM load
// 2D tensor_load_to_lds: f16 elements, tile = tile_d0 x tile_d1, row stride
// `stride` elements.  LDS padding: 1 DWORD after every 16 DWORDs (32 halves)
// -> LDS row stride 34 halves, bank-conflict-free fragment reads.
__device__ __forceinline__ void tdm_load_2d_f16(
    uint32_t lds_off, const _Float16* gptr,
    uint32_t tensor_d0, uint32_t tensor_d1, uint32_t stride,
    uint32_t tile_d0, uint32_t tile_d1) {
  uint64_t ga = (uint64_t)(uintptr_t)gptr;
  v4u g0;
  g0.x = 1u;                                                  // count=1
  g0.y = lds_off;                                             // LDS byte addr
  g0.z = (uint32_t)ga;                                        // global addr lo
  g0.w = (uint32_t)((ga >> 32) & 0x1FFFFFFu) | (2u << 30);    // addr hi | type=2
  v8i g1;
  g1[0] = (int)((1u << 16)        // data_size = 2 bytes
              | (1u << 20)        // pad_enable
              | (3u << 22)        // pad_interval code 3 = 16 DWORDs
              | (0u << 25));      // pad_amount  code 0 = 1 DWORD
  g1[1] = (int)((tensor_d0 & 0xFFFFu) << 16);
  g1[2] = (int)((tensor_d0 >> 16) | ((tensor_d1 & 0xFFFFu) << 16));
  g1[3] = (int)((tensor_d1 >> 16) | (tile_d0 << 16));
  g1[4] = (int)(tile_d1 & 0xFFFFu);                           // tile_dim2 = 0
  g1[5] = (int)stride;                                        // dim0_stride lo32
  g1[6] = 0;
  g1[7] = 0;
  v4i g2 = {0, 0, 0, 0};
  v4i g3 = {0, 0, 0, 0};
#if __clang_major__ >= 23
  v8i g4 = {0, 0, 0, 0, 0, 0, 0, 0};
  __builtin_amdgcn_tensor_load_to_lds(g0, g1, g2, g3, g4, 0);
#else
  __builtin_amdgcn_tensor_load_to_lds(g0, g1, g2, g3, 0);
#endif
}

// ---------------------------------------------------------------- LayerNorm
__global__ __launch_bounds__(256) void layernorm_k(
    const float* __restrict__ x, const float* __restrict__ g,
    const float* __restrict__ bta, float* __restrict__ y32,
    _Float16* __restrict__ y16) {
  const int t = blockIdx.x, tid = threadIdx.x;
  const int lane = tid & 31, wid = tid >> 5;
  const float* xr = x + (size_t)t * Dd;
  float4 f = *(const float4*)(xr + tid * 4);
  float s  = f.x + f.y + f.z + f.w;
  float ss = f.x*f.x + f.y*f.y + f.z*f.z + f.w*f.w;
  #pragma unroll
  for (int o = 16; o >= 1; o >>= 1) { s += __shfl_xor(s, o, 32); ss += __shfl_xor(ss, o, 32); }
  __shared__ float sh[16];
  if (lane == 0) { sh[wid] = s; sh[8 + wid] = ss; }
  __syncthreads();
  float S = 0.f, SS = 0.f;
  #pragma unroll
  for (int i = 0; i < 8; i++) { S += sh[i]; SS += sh[8 + i]; }
  const float mean = S / (float)Dd;
  const float var  = SS / (float)Dd - mean * mean;
  const float inv  = rsqrtf(var + 1e-5f);
  float4 gg = *(const float4*)(g + tid * 4);
  float4 bb = *(const float4*)(bta + tid * 4);
  float4 o;
  o.x = (f.x - mean) * inv * gg.x + bb.x;
  o.y = (f.y - mean) * inv * gg.y + bb.y;
  o.z = (f.z - mean) * inv * gg.z + bb.z;
  o.w = (f.w - mean) * inv * gg.w + bb.w;
  const size_t off = (size_t)t * Dd + tid * 4;
  *(float4*)(y32 + off) = o;
  y16[off + 0] = (_Float16)o.x; y16[off + 1] = (_Float16)o.y;
  y16[off + 2] = (_Float16)o.z; y16[off + 3] = (_Float16)o.w;
}

// --------------------------------------------- weight transpose + f32 -> f16
// W[z]: [K,N] f32 row-major  ->  Wt[z]: [N,K] f16 row-major
__global__ __launch_bounds__(256) void transpose_cvt_k(
    const float* __restrict__ W, _Float16* __restrict__ Wt,
    int K, int N, long sIn, long sOut) {
  __shared__ float tile[32][33];
  W  += (size_t)blockIdx.z * sIn;
  Wt += (size_t)blockIdx.z * sOut;
  const int n0 = blockIdx.x * 32, k0 = blockIdx.y * 32;
  const int tx = threadIdx.x & 31, ty = threadIdx.x >> 5;
  #pragma unroll
  for (int i = 0; i < 4; i++)
    tile[ty * 4 + i][tx] = W[(size_t)(k0 + ty * 4 + i) * N + n0 + tx];
  __syncthreads();
  #pragma unroll
  for (int i = 0; i < 4; i++)
    Wt[(size_t)(n0 + ty * 4 + i) * K + k0 + tx] = (_Float16)tile[tx][ty * 4 + i];
}

// -------------------------------------- WMMA GEMM fed by TDM (f16 operands)
// C[z] = act(A[z] @ Bt[z]^T + bias[z]) (+ res).  A:[M,K] f16, Bt:[N,K] f16.
// 128x128 block tile, K-step 32, double-buffered LDS filled by the TDM.
template <int OUT_F16, int HAS_BIAS, int ACT_GELU, int HAS_RES>
__global__ __launch_bounds__(256) void gemm_tdm(
    const _Float16* __restrict__ A, const _Float16* __restrict__ Bt,
    const float* __restrict__ bias, const float* __restrict__ res,
    void* __restrict__ Cout, int M, int N, int K,
    long sA, long sB, long sBias, long sC) {
  __shared__ _Float16 smem[2][2][128 * 34];   // [buf][A|B][row*34 + k]
  const int tid = threadIdx.x, lane = tid & 31, wid = tid >> 5;
  const int bm = blockIdx.y * 128, bn = blockIdx.x * 128;
  const int z = blockIdx.z;
  A  += (size_t)z * sA;
  Bt += (size_t)z * sB;
  if (HAS_BIAS) bias += (size_t)z * sBias;

  F32x8 acc[8];
  #pragma unroll
  for (int j = 0; j < 8; j++)
    #pragma unroll
    for (int i = 0; i < 8; i++) acc[j].f[i] = 0.f;

  const int nsteps = K >> 5;
  if (wid == 0) {
    tdm_load_2d_f16((uint32_t)(uintptr_t)&smem[0][0][0], A + (size_t)bm * K,
                    (uint32_t)K, (uint32_t)M, (uint32_t)K, 32u, 128u);
    tdm_load_2d_f16((uint32_t)(uintptr_t)&smem[0][1][0], Bt + (size_t)bn * K,
                    (uint32_t)K, (uint32_t)N, (uint32_t)K, 32u, 128u);
    __builtin_amdgcn_s_wait_tensorcnt(0);
  }
  __syncthreads();

  for (int s = 0; s < nsteps; s++) {
    const int cur = s & 1, nxt = cur ^ 1;
    if (wid == 0 && s + 1 < nsteps) {
      const int k0 = (s + 1) << 5;
      tdm_load_2d_f16((uint32_t)(uintptr_t)&smem[nxt][0][0],
                      A + (size_t)bm * K + k0,
                      (uint32_t)K, (uint32_t)M, (uint32_t)K, 32u, 128u);
      tdm_load_2d_f16((uint32_t)(uintptr_t)&smem[nxt][1][0],
                      Bt + (size_t)bn * K + k0,
                      (uint32_t)K, (uint32_t)N, (uint32_t)K, 32u, 128u);
    }
    const _Float16* al = &smem[cur][0][0];
    const _Float16* bl = &smem[cur][1][0];
    F16x16 af;
    #pragma unroll
    for (int p = 0; p < 8; p++) {
      const int kk = kmap(p, lane);
      af.u[p] = *(const uint32_t*)&al[(wid * 16 + (lane & 15)) * 34 + kk];
    }
    #pragma unroll
    for (int j = 0; j < 8; j++) {
      F16x16 bf;
      #pragma unroll
      for (int p = 0; p < 8; p++) {
        const int kk = kmap(p, lane);
        bf.u[p] = *(const uint32_t*)&bl[(j * 16 + (lane & 15)) * 34 + kk];
      }
      acc[j].v = wmma_f16(af.v, bf.v, acc[j].v);
    }
    if (wid == 0 && s + 1 < nsteps) __builtin_amdgcn_s_wait_tensorcnt(0);
    __syncthreads();
  }

  // epilogue
  #pragma unroll
  for (int r = 0; r < 8; r++) {
    const int row = bm + wid * 16 + r + ((lane >> 4) << 3);
    #pragma unroll
    for (int j = 0; j < 8; j++) {
      const int col = bn + j * 16 + (lane & 15);
      float v = acc[j].f[r];
      if (HAS_BIAS) v += bias[col];
      if (ACT_GELU) v = gelu_exact(v);
      if (HAS_RES)  v += res[(size_t)row * N + col];
      if (OUT_F16)
        ((_Float16*)Cout)[(size_t)z * sC + (size_t)row * N + col] = (_Float16)v;
      else
        ((float*)Cout)[(size_t)z * sC + (size_t)row * N + col] = v;
    }
  }
}

// ------------------------------------------- Flash attention (f16 in, WMMA)
__global__ __launch_bounds__(256) void flash_attn_k(
    const _Float16* __restrict__ q, const _Float16* __restrict__ k,
    const _Float16* __restrict__ v, _Float16* __restrict__ ctx) {
  __shared__ _Float16 plds[8 * 16 * 34];
  const int lane = threadIdx.x & 31, wid = threadIdx.x >> 5;
  const int b = blockIdx.z, h = blockIdx.y;
  const int q0 = (blockIdx.x * 8 + wid) * 16;
  _Float16* pl = plds + wid * 16 * 34;
  const size_t base = (size_t)b * Ss * Dd + (size_t)h * HD;

  // Q fragments (16 rows x 64 K -> two A frags), dword pair loads
  F16x16 qa[2];
  const _Float16* qrow = q + base + (size_t)(q0 + (lane & 15)) * Dd;
  #pragma unroll
  for (int f = 0; f < 2; f++)
    #pragma unroll
    for (int p = 0; p < 8; p++)
      qa[f].u[p] = *(const uint32_t*)&qrow[kmap(p, lane) + f * 32];

  F32x8 accO[4];
  #pragma unroll
  for (int dt = 0; dt < 4; dt++)
    #pragma unroll
    for (int i = 0; i < 8; i++) accO[dt].f[i] = 0.f;
  float mrun[8], lrun[8];
  #pragma unroll
  for (int r = 0; r < 8; r++) { mrun[r] = -1e30f; lrun[r] = 0.f; }

  const int ktmax = (q0 + 15) >> 5;
  for (int kt = 0; kt <= ktmax; kt++) {
    // scores: S = Q @ K^T (two 16-key column tiles)
    F32x8 sc[2];
    #pragma unroll
    for (int t = 0; t < 2; t++) {
      #pragma unroll
      for (int i = 0; i < 8; i++) sc[t].f[i] = 0.f;
      const int key = kt * 32 + t * 16 + (lane & 15);
      const _Float16* krow = k + base + (size_t)key * Dd;
      #pragma unroll
      for (int f = 0; f < 2; f++) {
        F16x16 bf;
        #pragma unroll
        for (int p = 0; p < 8; p++)
          bf.u[p] = *(const uint32_t*)&krow[kmap(p, lane) + f * 32];
        sc[t].v = wmma_f16(qa[f].v, bf.v, sc[t].v);
      }
    }
    // causal mask + online softmax
    const int rbase = q0 + ((lane >> 4) << 3);
    #pragma unroll
    for (int r = 0; r < 8; r++) {
      float s0 = sc[0].f[r] * SCALE_INV;
      float s1 = sc[1].f[r] * SCALE_INV;
      const int key0 = kt * 32 + (lane & 15);
      if (key0 > rbase + r)      s0 = -1e30f;
      if (key0 + 16 > rbase + r) s1 = -1e30f;
      float mx = fmaxf(s0, s1);
      #pragma unroll
      for (int o = 8; o >= 1; o >>= 1) mx = fmaxf(mx, __shfl_xor(mx, o, 32));
      const float mnew = fmaxf(mrun[r], mx);
      s0 = __expf(s0 - mnew);
      s1 = __expf(s1 - mnew);
      float rs = s0 + s1;
      #pragma unroll
      for (int o = 8; o >= 1; o >>= 1) rs += __shfl_xor(rs, o, 32);
      const float alpha = __expf(mrun[r] - mnew);
      lrun[r] = lrun[r] * alpha + rs;
      mrun[r] = mnew;
      #pragma unroll
      for (int dt = 0; dt < 4; dt++) accO[dt].f[r] *= alpha;
      const int rowl = r + ((lane >> 4) << 3);
      pl[rowl * 34 + (lane & 15)]      = (_Float16)s0;
      pl[rowl * 34 + 16 + (lane & 15)] = (_Float16)s1;
    }
    asm volatile("s_wait_dscnt 0" ::: "memory");
    // P in A-fragment layout (16 rows x 32 keys)
    F16x16 pa;
    #pragma unroll
    for (int p = 0; p < 8; p++)
      pa.u[p] = *(const uint32_t*)&pl[(lane & 15) * 34 + kmap(p, lane)];
    // O += P @ V
    #pragma unroll
    for (int dt = 0; dt < 4; dt++) {
      F16x16 vb;
      #pragma unroll
      for (int p = 0; p < 8; p++) {
        const int key = kt * 32 + kmap(p, lane);
        vb.h[2 * p]     = v[base + (size_t)key * Dd + dt * 16 + (lane & 15)];
        vb.h[2 * p + 1] = v[base + (size_t)(key + 1) * Dd + dt * 16 + (lane & 15)];
      }
      accO[dt].v = wmma_f16(pa.v, vb.v, accO[dt].v);
    }
    asm volatile("" ::: "memory");
  }
  #pragma unroll
  for (int r = 0; r < 8; r++) {
    const float inv = 1.f / lrun[r];
    const int row = q0 + r + ((lane >> 4) << 3);
    #pragma unroll
    for (int dt = 0; dt < 4; dt++)
      ctx[base + (size_t)row * Dd + dt * 16 + (lane & 15)] =
          (_Float16)(accO[dt].f[r] * inv);
  }
}

// -------------------------------------------------------------------- Gating
__global__ __launch_bounds__(256) void gate_k(
    const float* __restrict__ xln2, const float* __restrict__ Wg,
    const float* __restrict__ bg, int* __restrict__ route,
    float* __restrict__ prob_sum) {
  const int lane = threadIdx.x & 31, wid = threadIdx.x >> 5;
  const int t = blockIdx.x * 8 + wid;
  const float* xr = xln2 + (size_t)t * Dd;
  float a[8] = {0, 0, 0, 0, 0, 0, 0, 0};
  for (int d = lane; d < Dd; d += 32) {
    const float xv = xr[d];
    float4 w0 = *(const float4*)(Wg + d * 8);
    float4 w1 = *(const float4*)(Wg + d * 8 + 4);
    a[0] += xv * w0.x; a[1] += xv * w0.y; a[2] += xv * w0.z; a[3] += xv * w0.w;
    a[4] += xv * w1.x; a[5] += xv * w1.y; a[6] += xv * w1.z; a[7] += xv * w1.w;
  }
  #pragma unroll
  for (int e = 0; e < 8; e++) {
    #pragma unroll
    for (int o = 16; o >= 1; o >>= 1) a[e] += __shfl_xor(a[e], o, 32);
    a[e] += bg[e];
  }
  float mx = a[0]; int arg = 0;
  #pragma unroll
  for (int e = 1; e < 8; e++) if (a[e] > mx) { mx = a[e]; arg = e; }
  float pr[8], se = 0.f;
  #pragma unroll
  for (int e = 0; e < 8; e++) { pr[e] = __expf(a[e] - mx); se += pr[e]; }
  const float inv = 1.f / se;
  if (lane == 0) {
    route[t] = arg;
    #pragma unroll
    for (int e = 0; e < 8; e++) atomicAdd(&prob_sum[e], pr[e] * inv);
  }
}

// -------------------------------- Serial routing scan (wave32 ballot prefix)
__global__ void scan_route_k(const int* __restrict__ route, int* __restrict__ slot,
                             int* __restrict__ keep, float* __restrict__ counts,
                             float* __restrict__ ndrop) {
  const int lane = threadIdx.x;
  int base[8] = {0, 0, 0, 0, 0, 0, 0, 0};
  int kept = 0;
  const unsigned lt = (1u << lane) - 1u;
  for (int t0 = 0; t0 < Tt; t0 += 32) {
    const int r = route[t0 + lane];
    int pos = 0;
    #pragma unroll
    for (int e = 0; e < 8; e++) {
      unsigned m = (unsigned)__ballot(r == e);
      if (r == e) pos = base[e] + __popc(m & lt);
      base[e] += __popc(m);
    }
    const int kp = (pos < CAPc) ? 1 : 0;
    keep[t0 + lane] = kp;
    slot[t0 + lane] = r * CAPc + pos;
    kept += __popc((unsigned)__ballot(kp));
  }
  if (lane == 0) ndrop[0] = (float)(Tt - kept);
  if (lane < 8) counts[lane] = (float)base[lane];
}

// ---------------------------------------------------------- Dispatch/Combine
__global__ __launch_bounds__(256) void dispatch_k(
    const float* __restrict__ xln2, const int* __restrict__ slot,
    const int* __restrict__ keep, _Float16* __restrict__ buf) {
  const int t = blockIdx.x;
  if (!keep[t]) return;
  const int s = slot[t];
  float4 f = *(const float4*)(xln2 + (size_t)t * Dd + threadIdx.x * 4);
  _Float16* o = buf + (size_t)s * Dd + threadIdx.x * 4;
  o[0] = (_Float16)f.x; o[1] = (_Float16)f.y;
  o[2] = (_Float16)f.z; o[3] = (_Float16)f.w;
}

__global__ __launch_bounds__(256) void combine_k(
    const float* __restrict__ xattn, const float* __restrict__ xln2,
    const float* __restrict__ eo, const int* __restrict__ slot,
    const int* __restrict__ keep, float* __restrict__ out) {
  const int t = blockIdx.x;
  const size_t o = (size_t)t * Dd + threadIdx.x * 4;
  float4 a = *(const float4*)(xattn + o);
  float4 c;
  if (keep[t]) c = *(const float4*)(eo + (size_t)slot[t] * Dd + threadIdx.x * 4);
  else         c = *(const float4*)(xln2 + o);
  a.x += c.x; a.y += c.y; a.z += c.z; a.w += c.w;
  *(float4*)(out + o) = a;
}

__global__ void zero_tail_k(float* p) { if (threadIdx.x < 17) p[threadIdx.x] = 0.f; }

// --------------------------------------------------------------------- Host
extern "C" void kernel_launch(void* const* d_in, const int* in_sizes, int n_in,
                              void* d_out, int out_size, void* d_ws, size_t ws_size,
                              hipStream_t stream) {
  const float* x    = (const float*)d_in[0];
  const float* Wq   = (const float*)d_in[2];
  const float* Wk   = (const float*)d_in[3];
  const float* Wv   = (const float*)d_in[4];
  const float* Wo   = (const float*)d_in[5];
  const float* ln1g = (const float*)d_in[6];
  const float* ln1b = (const float*)d_in[7];
  const float* ln2g = (const float*)d_in[8];
  const float* ln2b = (const float*)d_in[9];
  const float* Wg   = (const float*)d_in[10];
  const float* bg   = (const float*)d_in[11];
  const float* W1   = (const float*)d_in[12];
  const float* b1   = (const float*)d_in[13];
  const float* W2   = (const float*)d_in[14];
  const float* b2   = (const float*)d_in[15];

  float* out    = (float*)d_out;
  float* counts = out + (size_t)Tt * Dd;
  float* psum   = counts + 8;
  float* ndrop  = psum + 8;

  // ---- workspace arena
  const size_t TD  = (size_t)Tt * Dd;        // 8388608 elements
  const size_t WDD = (size_t)Dd * Dd;        // 1048576
  const size_t WDF = (size_t)Ee * Dd * FFf;  // 33554432
  char* w = (char*)d_ws;
  auto alloc = [&](size_t bytes) { char* p = w; w += (bytes + 255) & ~(size_t)255; return p; };

  _Float16* Wqt  = (_Float16*)alloc(WDD * 2);
  _Float16* Wkt  = (_Float16*)alloc(WDD * 2);
  _Float16* Wvt  = (_Float16*)alloc(WDD * 2);
  _Float16* Wot  = (_Float16*)alloc(WDD * 2);
  _Float16* W1t  = (_Float16*)alloc(WDF * 2);   // [E][FF,D]
  _Float16* W2t  = (_Float16*)alloc(WDF * 2);   // [E][D,FF]
  _Float16* xln1h = (_Float16*)alloc(TD * 2);   // later overlaid by hbuf
  _Float16* qh    = (_Float16*)alloc(TD * 2);
  _Float16* kh    = (_Float16*)alloc(TD * 2);
  _Float16* vh    = (_Float16*)alloc(TD * 2);
  _Float16* ctxh  = (_Float16*)alloc(TD * 2);   // later overlaid by bufh
  float* xattn = (float*)alloc(TD * 4);         // also ln1 f32 scratch
  float* xln2  = (float*)alloc(TD * 4);
  float* eo    = (float*)alloc(TD * 4);
  int* route = (int*)alloc(Tt * 4);
  int* slot  = (int*)alloc(Tt * 4);
  int* keep  = (int*)alloc(Tt * 4);
  _Float16* hbuf = xln1h;                       // [E*CAP, FF] = 4*TD halves
  _Float16* bufh = ctxh;                        // [E*CAP, D]
  _Float16* ln2h = ctxh;                        // dead f16 sink for ln2

  zero_tail_k<<<1, 32, 0, stream>>>(counts);

  // weights -> transposed f16 (resident in 192MB L2 thereafter)
  transpose_cvt_k<<<dim3(32, 32, 1), 256, 0, stream>>>(Wq, Wqt, Dd, Dd, 0, 0);
  transpose_cvt_k<<<dim3(32, 32, 1), 256, 0, stream>>>(Wk, Wkt, Dd, Dd, 0, 0);
  transpose_cvt_k<<<dim3(32, 32, 1), 256, 0, stream>>>(Wv, Wvt, Dd, Dd, 0, 0);
  transpose_cvt_k<<<dim3(32, 32, 1), 256, 0, stream>>>(Wo, Wot, Dd, Dd, 0, 0);
  transpose_cvt_k<<<dim3(FFf / 32, Dd / 32, Ee), 256, 0, stream>>>(
      W1, W1t, Dd, FFf, (long)Dd * FFf, (long)FFf * Dd);
  transpose_cvt_k<<<dim3(Dd / 32, FFf / 32, Ee), 256, 0, stream>>>(
      W2, W2t, FFf, Dd, (long)FFf * Dd, (long)Dd * FFf);

  layernorm_k<<<Tt, 256, 0, stream>>>(x, ln1g, ln1b, xattn, xln1h);

  dim3 gproj(Dd / 128, Tt / 128, 1);
  gemm_tdm<1, 0, 0, 0><<<gproj, 256, 0, stream>>>(xln1h, Wqt, nullptr, nullptr,
                                                  qh, Tt, Dd, Dd, 0, 0, 0, 0);
  gemm_tdm<1, 0, 0, 0><<<gproj, 256, 0, stream>>>(xln1h, Wkt, nullptr, nullptr,
                                                  kh, Tt, Dd, Dd, 0, 0, 0, 0);
  gemm_tdm<1, 0, 0, 0><<<gproj, 256, 0, stream>>>(xln1h, Wvt, nullptr, nullptr,
                                                  vh, Tt, Dd, Dd, 0, 0, 0, 0);

  flash_attn_k<<<dim3(Ss / 128, Hh, Bb), 256, 0, stream>>>(qh, kh, vh, ctxh);

  gemm_tdm<0, 0, 0, 1><<<gproj, 256, 0, stream>>>(ctxh, Wot, nullptr, x,
                                                  xattn, Tt, Dd, Dd, 0, 0, 0, 0);

  layernorm_k<<<Tt, 256, 0, stream>>>(xattn, ln2g, ln2b, xln2, ln2h);

  gate_k<<<Tt / 8, 256, 0, stream>>>(xln2, Wg, bg, route, psum);
  scan_route_k<<<1, 32, 0, stream>>>(route, slot, keep, counts, ndrop);
  dispatch_k<<<Tt, 256, 0, stream>>>(xln2, slot, keep, bufh);

  gemm_tdm<1, 1, 1, 0><<<dim3(FFf / 128, CAPc / 128, Ee), 256, 0, stream>>>(
      bufh, W1t, b1, nullptr, hbuf, CAPc, FFf, Dd,
      (long)CAPc * Dd, (long)FFf * Dd, FFf, (long)CAPc * FFf);

  gemm_tdm<0, 1, 0, 0><<<dim3(Dd / 128, CAPc / 128, Ee), 256, 0, stream>>>(
      hbuf, W2t, b2, nullptr, eo, CAPc, Dd, FFf,
      (long)CAPc * FFf, (long)Dd * FFf, Dd, (long)CAPc * Dd);

  combine_k<<<Tt, 256, 0, stream>>>(xattn, xln2, eo, slot, keep, out);

  (void)in_sizes; (void)n_in; (void)out_size; (void)ws_size;
}
